// GCN2ReLU_53197464928899
// MI455X (gfx1250) — matrix operand order repaired and checked by reading, not compiled, not run on hardware
//
#include <hip/hip_runtime.h>
#include <cmath>
#include <cstdint>
#include <cstddef>

// ---------------------------------------------------------------------------
// GCNII (GCN2+ReLU) pipeline for MI455X / gfx1250, wave32.
//  - CSR-by-dst built once per launch (no per-layer scatter atomics)
//  - all node-feature GEMMs via v_wmma_f32_16x16x32_f16 (f32 accumulate)
//  - aggregation is L2-resident gather (h = 51.2MB << 192MB L2) with
//    shuffle-broadcast edge metadata
// ---------------------------------------------------------------------------

typedef __attribute__((ext_vector_type(16))) _Float16 v16h;
typedef __attribute__((ext_vector_type(8)))  float    v8f;

#define HID   128
#define NGRPH 512
#define OUTC  10

struct U4 { uint32_t x, y, z, w; };
struct U2 { uint32_t x, y; };
struct F4 { float x, y, z, w; };

union ABFrag { v16h v; U4 q[2]; };
union CFrag  { v8f  v; float f[8]; };
union H2pack { _Float16 h[2]; uint32_t u; };
union H4pack { _Float16 h[4]; U2 u; };

// ---------------------------------------------------------------- utilities
__global__ __launch_bounds__(256) void k_zero_u32(uint32_t* __restrict__ p, int n) {
  int i = blockIdx.x * 256 + threadIdx.x;
  if (i < n) p[i] = 0u;
}

__global__ __launch_bounds__(256) void k_f32_to_f16(const float* __restrict__ src,
                                                    _Float16* __restrict__ dst, int n) {
  int i = blockIdx.x * 256 + threadIdx.x;
  if (i < n) dst[i] = (_Float16)src[i];
}

// ------------------------------------------------------------ degree / CSR
__global__ __launch_bounds__(256) void k_count(const int* __restrict__ ei, int E,
                                               int* __restrict__ counts) {
  int e = blockIdx.x * 256 + threadIdx.x;
  if (e < E) atomicAdd(&counts[ei[E + e]], 1);
}

__global__ __launch_bounds__(256) void k_dinv(const int* __restrict__ counts,
                                              float* __restrict__ dinv, int n) {
  int i = blockIdx.x * 256 + threadIdx.x;
  if (i < n) dinv[i] = rsqrtf((float)(counts[i] + 1));  // +1 self loop, deg >= 1
}

// exclusive scan, stage 1: per 256-block
__global__ __launch_bounds__(256) void k_scan_block(const int* __restrict__ counts,
                                                    int* __restrict__ excl,
                                                    int* __restrict__ bsums, int n) {
  __shared__ int sh[256];
  int i = blockIdx.x * 256 + threadIdx.x;
  int v = (i < n) ? counts[i] : 0;
  sh[threadIdx.x] = v;
  __syncthreads();
  for (int off = 1; off < 256; off <<= 1) {
    int add = (threadIdx.x >= off) ? sh[threadIdx.x - off] : 0;
    __syncthreads();
    sh[threadIdx.x] += add;
    __syncthreads();
  }
  if (i < n) excl[i] = sh[threadIdx.x] - v;
  if (threadIdx.x == 255) bsums[blockIdx.x] = sh[255];
}

// exclusive scan, stage 2: single block over block sums (nb <= 512)
__global__ __launch_bounds__(512) void k_scan_top(int* __restrict__ bs, int nb) {
  __shared__ int sh[512];
  int t = threadIdx.x;
  int v = (t < nb) ? bs[t] : 0;
  sh[t] = v;
  __syncthreads();
  for (int off = 1; off < 512; off <<= 1) {
    int add = (t >= off) ? sh[t - off] : 0;
    __syncthreads();
    sh[t] += add;
    __syncthreads();
  }
  if (t < nb) bs[t] = sh[t] - v;
}

__global__ __launch_bounds__(256) void k_finalize_rowptr(const int* __restrict__ excl,
                                                         const int* __restrict__ boffs,
                                                         int* __restrict__ row_ptr,
                                                         int* __restrict__ cursor,
                                                         int n, int E) {
  int i = blockIdx.x * 256 + threadIdx.x;
  if (i < n) {
    int v = excl[i] + boffs[i >> 8];
    row_ptr[i] = v;
    cursor[i] = v;
  }
  if (i == 0) row_ptr[n] = E;
}

__global__ __launch_bounds__(256) void k_fill(const int* __restrict__ ei, int E,
                                              int* __restrict__ cursor,
                                              int* __restrict__ col,
                                              float* __restrict__ val,
                                              const float* __restrict__ dinv) {
  int e = blockIdx.x * 256 + threadIdx.x;
  if (e >= E) return;
  int src = ei[e];
  int dst = ei[E + e];
  int pos = atomicAdd(&cursor[dst], 1);
  col[pos] = src;
  val[pos] = dinv[src];
}

// -------------------------------------------- weight pre-swizzle (B fragments)
// B-fragment layout (16-bit 32x16): lane = 16*khalf + n16; dword v holds
// B[kb][n], B[kb + 1][n] with kb = (v<4) ? khalf*8 + 2v : 16 + khalf*8 + 2(v-4).  Stored as
// wfrag[(((mat*8 + ctile)*4 + ktile)*32 + lane)*8 + v].
// mat 0 = lin0 (B[k][n] = lin0_w[n][k]); mat 1..8 = conv_w[l] (B[k][n] = W[k][n]).
__global__ __launch_bounds__(256) void k_build_wfrag(const float* __restrict__ lin0_w,
                                                     const float* __restrict__ conv_w,
                                                     uint32_t* __restrict__ wfrag) {
  int idx = blockIdx.x * 256 + threadIdx.x;
  const int total = 9 * 8 * 4 * 32 * 8;
  if (idx >= total) return;
  int v     = idx & 7;
  int lane  = (idx >> 3) & 31;
  int kt    = (idx >> 8) & 3;
  int ct    = (idx >> 10) & 7;
  int mat   = idx >> 13;
  int n     = ct * 16 + (lane & 15);
  int khalf = lane >> 4;
  int k     = kt * 32 + ((v & 4) << 2) + khalf * 8 + (v & 3) * 2;
  float f0, f1;
  if (mat == 0) {
    f0 = lin0_w[n * HID + k];
    f1 = lin0_w[n * HID + k + 1];
  } else {
    const float* W = conv_w + (size_t)(mat - 1) * HID * HID;
    f0 = W[k * HID + n];
    f1 = W[(k + 1) * HID + n];
  }
  H2pack p;
  p.h[0] = (_Float16)f0;
  p.h[1] = (_Float16)f1;
  wfrag[idx] = p.u;
}

// ---------------------------------------------------------- WMMA GEMM kernel
// MODE 0 (lin0):  Out = Out2 = relu(A @ B + bias)
// MODE 1 (layer): Out = relu(dscale*(A @ B) + cscale*dscale*Cadd)
//                 with cscale=(1-beta)/beta, dscale=beta
//                 i.e. relu(beta*(s@W) + (1-beta)*s); residual rides in the
//                 WMMA C operand so the epilogue is one scale+relu.
// One wave per 16x16 tile, 8 waves/block cover the 128-wide strip;
// K=128 as 4 x v_wmma_f32_16x16x32_f16.
template <int MODE>
__global__ __launch_bounds__(256) void k_gemm_wmma(const _Float16* __restrict__ A,
                                                   const uint32_t* __restrict__ Bfrag,
                                                   const float* __restrict__ Cadd,
                                                   const float* __restrict__ bias,
                                                   float* __restrict__ Out,
                                                   float* __restrict__ Out2,
                                                   float cscale, float dscale, int M) {
  const int wave = threadIdx.x >> 5;
  const int lane = threadIdx.x & 31;
  const int half = lane >> 4;
  const int l16  = lane & 15;
  const int m0   = blockIdx.x << 4;
  if (m0 >= M) return;
  const int n0 = wave << 4;
  const int mr = m0 + l16;  // A row owned by this lane
  const int nc = n0 + l16;  // output column owned by this lane
  // base element offset for this lane's C/D column: rows m0+8*half .. +7, col nc
  const size_t cbase = (size_t)(m0 + 8 * half) * HID + nc;

  CFrag c;
  if (MODE == 0) {
    const float bv = bias[nc];
#pragma unroll
    for (int v = 0; v < 8; ++v) c.f[v] = bv;
  } else {
#pragma unroll
    for (int v = 0; v < 8; ++v) c.f[v] = Cadd[cbase + (size_t)v * HID] * cscale;
  }
  v8f acc = c.v;

  const _Float16* arow = A + (size_t)mr * HID + half * 8;
#pragma unroll
  for (int kt = 0; kt < 4; ++kt) {
    ABFrag a, b;
    // 16-bit A 16x32 fragment: two contiguous 16B loads per lane.
    a.q[0] = *(const U4*)(arow + kt * 32);
    a.q[1] = *(const U4*)(arow + kt * 32 + 16);
    const uint32_t* bp = Bfrag + (size_t)(((wave * 4 + kt) * 32 + lane) << 3);
    b.q[0] = *(const U4*)bp;
    b.q[1] = *(const U4*)(bp + 4);
    acc = __builtin_amdgcn_wmma_f32_16x16x32_f16(
        false, a.v, false, b.v, (short)0, acc, false, false);
  }

  c.v = acc;
#pragma unroll
  for (int v = 0; v < 8; ++v) {
    float r = (MODE == 0) ? fmaxf(c.f[v], 0.0f)
                          : fmaxf(c.f[v] * dscale, 0.0f);
    Out[cbase + (size_t)v * HID] = r;
    if (MODE == 0) Out2[cbase + (size_t)v * HID] = r;
  }
}

// -------------------------------------------- fused aggregate + residual mix
// One wave per dst row; lane owns 4 features (float4).  Edge metadata is
// loaded coalesced (one edge per lane) and broadcast with __shfl, so the
// hot loop does exactly one 16B h-row load per lane per edge (L2-resident).
// s = 0.9*(dinv[r]*(sum val[e]*h[src]) + dinv[r]^2*h[r]) + 0.1*h0[r]
// Writes s (f32, WMMA C operand) and s_f16 (WMMA A operand).
__global__ __launch_bounds__(256) void k_agg_combine(const int* __restrict__ row_ptr,
                                                     const int* __restrict__ col,
                                                     const float* __restrict__ val,
                                                     const float* __restrict__ dinv,
                                                     const float* __restrict__ h,
                                                     const float* __restrict__ h0,
                                                     float* __restrict__ s,
                                                     _Float16* __restrict__ s16,
                                                     int nrows) {
  const int wave = threadIdx.x >> 5;
  const int lane = threadIdx.x & 31;
  const int row  = blockIdx.x * 8 + wave;
  if (row >= nrows) return;
  const int c4  = lane << 2;
  const int beg = row_ptr[row];
  const int end = row_ptr[row + 1];
  float a0 = 0.f, a1 = 0.f, a2 = 0.f, a3 = 0.f;
  for (int base = beg; base < end; base += 32) {
    const int nedge = end - base;               // >0
    const int eidx  = base + lane;
    const int   mySrc = (lane < nedge) ? col[eidx] : 0;
    const float myW   = (lane < nedge) ? val[eidx] : 0.0f;
    const int lim = (nedge < 32) ? nedge : 32;
    for (int j = 0; j < lim; ++j) {
      const int   src = __shfl(mySrc, j, 32);
      const float w   = __shfl(myW, j, 32);
      F4 hv = *(const F4*)(h + (size_t)src * HID + c4);
      a0 = fmaf(w, hv.x, a0);
      a1 = fmaf(w, hv.y, a1);
      a2 = fmaf(w, hv.z, a2);
      a3 = fmaf(w, hv.w, a3);
    }
  }
  const float dr = dinv[row];
  F4 hr = *(const F4*)(h  + (size_t)row * HID + c4);
  F4 hz = *(const F4*)(h0 + (size_t)row * HID + c4);
  F4 so;
  so.x = 0.9f * dr * (a0 + dr * hr.x) + 0.1f * hz.x;
  so.y = 0.9f * dr * (a1 + dr * hr.y) + 0.1f * hz.y;
  so.z = 0.9f * dr * (a2 + dr * hr.z) + 0.1f * hz.z;
  so.w = 0.9f * dr * (a3 + dr * hr.w) + 0.1f * hz.w;
  *(F4*)(s + (size_t)row * HID + c4) = so;
  H4pack p;
  p.h[0] = (_Float16)so.x;
  p.h[1] = (_Float16)so.y;
  p.h[2] = (_Float16)so.z;
  p.h[3] = (_Float16)so.w;
  *(U2*)(s16 + (size_t)row * HID + c4) = p.u;
}

// ----------------------------------------------------------------- pooling
__global__ __launch_bounds__(256) void k_pool_feat(const float* __restrict__ h,
                                                   const int* __restrict__ batch,
                                                   float* __restrict__ gsum, int total) {
  int i = blockIdx.x * 256 + threadIdx.x;
  if (i >= total) return;
  int n = i >> 7;
  int c = i & 127;
  atomicAdd(&gsum[batch[n] * HID + c], h[i]);
}

__global__ __launch_bounds__(256) void k_pool_cnt(const int* __restrict__ batch,
                                                  float* __restrict__ gcnt, int n) {
  int i = blockIdx.x * 256 + threadIdx.x;
  if (i < n) atomicAdd(&gcnt[batch[i]], 1.0f);
}

__global__ __launch_bounds__(256) void k_head(const float* __restrict__ gsum,
                                              const float* __restrict__ gcnt,
                                              const float* __restrict__ w,
                                              const float* __restrict__ b,
                                              float* __restrict__ out) {
  int i = blockIdx.x * 256 + threadIdx.x;
  if (i >= NGRPH * OUTC) return;
  int g = i / OUTC;
  int o = i - g * OUTC;
  float inv = 1.0f / fmaxf(gcnt[g], 1.0f);
  float acc = 0.0f;
  const float* gp = gsum + (size_t)g * HID;
  const float* wp = w + (size_t)o * HID;
#pragma unroll 8
  for (int c = 0; c < HID; ++c) acc = fmaf(gp[c], wp[c], acc);
  out[i] = acc * inv + b[o];
}

// ---------------------------------------------------------------------------
static inline size_t alignUp(size_t v, size_t a) { return (v + a - 1) / a * a; }

extern "C" void kernel_launch(void* const* d_in, const int* in_sizes, int n_in,
                              void* d_out, int out_size, void* d_ws, size_t ws_size,
                              hipStream_t stream) {
  const float* x      = (const float*)d_in[0];
  const int*   ei     = (const int*)d_in[1];
  const int*   batch  = (const int*)d_in[3];
  const float* lin0_w = (const float*)d_in[4];
  const float* lin0_b = (const float*)d_in[5];
  const float* lin1_w = (const float*)d_in[6];
  const float* lin1_b = (const float*)d_in[7];
  const float* conv_w = (const float*)d_in[8];
  float* out = (float*)d_out;

  const int N = in_sizes[0] / HID;  // 100000
  const int E = in_sizes[1] / 2;    // 1600000

  // ---- carve workspace
  char* p = (char*)d_ws;
  auto take = [&](size_t bytes) -> void* {
    void* r = (void*)p;
    p += alignUp(bytes, 256);
    return r;
  };
  int*       counts  = (int*)take((size_t)N * 4);
  int*       excl    = (int*)take((size_t)N * 4);
  int*       boffs   = (int*)take(1024 * 4);
  int*       row_ptr = (int*)take((size_t)(N + 1) * 4);
  int*       cursor  = (int*)take((size_t)N * 4);
  int*       col     = (int*)take((size_t)E * 4);
  float*     val     = (float*)take((size_t)E * 4);
  float*     dinv    = (float*)take((size_t)N * 4);
  uint32_t*  wfrag   = (uint32_t*)take((size_t)9 * 8 * 4 * 32 * 8 * 4);
  _Float16*  a16     = (_Float16*)take((size_t)N * HID * 2);
  float*     h       = (float*)take((size_t)N * HID * 4);
  float*     h0      = (float*)take((size_t)N * HID * 4);
  float*     s       = (float*)take((size_t)N * HID * 4);
  float*     gsum    = (float*)take((size_t)NGRPH * HID * 4);
  float*     gcnt    = (float*)take((size_t)NGRPH * 4);

  const int bN  = (N + 255) / 256;
  const int bE  = (E + 255) / 256;
  const int bNH = (N * HID + 255) / 256;
  const int nb  = bN;  // scan stage-1 block count (391 <= 512)

  // ---- CSR by dst + gcn norm
  k_zero_u32<<<bN, 256, 0, stream>>>((uint32_t*)counts, N);
  k_count<<<bE, 256, 0, stream>>>(ei, E, counts);
  k_dinv<<<bN, 256, 0, stream>>>(counts, dinv, N);
  k_scan_block<<<nb, 256, 0, stream>>>(counts, excl, boffs, N);
  k_scan_top<<<1, 512, 0, stream>>>(boffs, nb);
  k_finalize_rowptr<<<bN, 256, 0, stream>>>(excl, boffs, row_ptr, cursor, N, E);
  k_fill<<<bE, 256, 0, stream>>>(ei, E, cursor, col, val, dinv);

  // ---- weights -> swizzled fp16 B fragments
  k_build_wfrag<<<(9 * 8 * 4 * 32 * 8 + 255) / 256, 256, 0, stream>>>(lin0_w, conv_w, wfrag);

  // ---- h = h0 = relu(x @ lin0_w^T + b) via WMMA
  k_f32_to_f16<<<bNH, 256, 0, stream>>>(x, a16, N * HID);
  k_gemm_wmma<0><<<(N + 15) / 16, 256, 0, stream>>>(a16, wfrag, nullptr, lin0_b,
                                                    h, h0, 0.0f, 1.0f, N);

  // ---- 8 GCNII layers
  for (int l = 0; l < 8; ++l) {
    const float beta = logf(0.5f / (float)(l + 1) + 1.0f);
    k_agg_combine<<<(N + 7) / 8, 256, 0, stream>>>(row_ptr, col, val, dinv,
                                                   h, h0, s, a16, N);
    k_gemm_wmma<1><<<(N + 15) / 16, 256, 0, stream>>>(
        a16, wfrag + (size_t)(l + 1) * 8 * 4 * 32 * 8, s, nullptr,
        h, nullptr, (1.0f - beta) / beta, beta, N);
  }

  // ---- global mean pool + head
  k_zero_u32<<<(NGRPH * HID + 255) / 256, 256, 0, stream>>>((uint32_t*)gsum, NGRPH * HID);
  k_zero_u32<<<(NGRPH + 255) / 256, 256, 0, stream>>>((uint32_t*)gcnt, NGRPH);
  k_pool_feat<<<bNH, 256, 0, stream>>>(h, batch, gsum, N * HID);
  k_pool_cnt<<<bN, 256, 0, stream>>>(batch, gcnt, N);
  k_head<<<(NGRPH * OUTC + 255) / 256, 256, 0, stream>>>(gsum, gcnt, lin1_w, lin1_b, out);

  (void)n_in; (void)out_size; (void)ws_size;
}
